// DIModel_38311108280787
// MI455X (gfx1250) — compile-verified, matrix operand order checked by reading, not run on hardware
//
#include <hip/hip_runtime.h>
#include <hip/hip_bf16.h>

// ---------------------------------------------------------------------------
// MI455X (gfx1250, wave32) implementation.
// Heavy convs + FCs run as f16 WMMA GEMMs (v_wmma_f32_16x16x32_f16): the f32
// WMMA path is K=4/instr (8x lower matrix throughput), and with the whole
// working set resident in the 192MB L2 the big convs are compute-bound, so
// f16 inputs / f32 accumulation is the chip-rewarded precision.
//
// Convs are tap-major: conv(k=11) = sum of 11 channel-GEMMs with shifted
// inputs, so the LDS im2col gather uses compile-time immediate DS offsets
// (no per-element address arithmetic in the WMMA loop).
// ---------------------------------------------------------------------------

typedef __attribute__((ext_vector_type(16))) _Float16 v16h;
typedef __attribute__((ext_vector_type(8)))  _Float16 v8h;
typedef __attribute__((ext_vector_type(8)))  float    v8f;

union V16 { v16h v; v8h h[2]; };

__device__ __forceinline__ v8f wmma32f16(v16h a, v16h b, v8f c) {
  // D = A(16x32 f16) * B(32x16 f16) + C(16x16 f32)
  return __builtin_amdgcn_wmma_f32_16x16x32_f16(false, a, false, b, (short)0, c,
                                                false, false);
}

// Load a 16x32 f16 fragment from row-major [16 rows][stride] memory.
// CDNA5 layout: lane holds row (lane&15); element e holds
// K = ((e>>3)<<4) + ((lane>>4)<<3) + (e&7)  -> two contiguous 16B chunks/lane.
__device__ __forceinline__ v16h frag_ld(const _Float16* rowbase, size_t stride,
                                        int lane) {
  const _Float16* p = rowbase + (size_t)(lane & 15) * stride + ((lane >> 4) << 3);
  V16 r;
  r.h[0] = *(const v8h*)(p);
  r.h[1] = *(const v8h*)(p + 16);
  return r.v;
}

// ------------------------------ FC (NT GEMM) -------------------------------
// Y[M,N] = act(X[M,K] * W[N,K]^T + bias[N]).  K multiple of 32, M mult 32,
// N mult 64 per grid.  Block = 8 waves covering 32(M) x 64(N).
template <int ACT> // 0 none, 1 relu, 2 sigmoid
__global__ __launch_bounds__(256) void fc_wmma(
    const _Float16* __restrict__ X, const _Float16* __restrict__ Wt,
    const float* __restrict__ bias, _Float16* __restrict__ Yh,
    float* __restrict__ Yf, int M, int N, int K) {
  const int tid = threadIdx.x;
  const int lane = tid & 31;
  const int wave = tid >> 5;
  const int m0 = blockIdx.y * 32 + (wave & 1) * 16;
  const int n0 = blockIdx.x * 64 + (wave >> 1) * 16;
  const _Float16* Xr = X + (size_t)m0 * K;
  const _Float16* Wr = Wt + (size_t)n0 * K;
  v8f acc = {};
  for (int k0 = 0; k0 < K; k0 += 32) {
    __builtin_prefetch(Xr + k0 + 256, 0, 1);   // global_prefetch_b8
    __builtin_prefetch(Wr + k0 + 256, 0, 1);
    v16h a = frag_ld(Xr + k0, (size_t)K, lane);
    v16h b = frag_ld(Wr + k0, (size_t)K, lane);
    acc = wmma32f16(a, b, acc);
  }
  // D layout: VGPR r -> row m0 + r + 8*(lane>=16); col = n0 + (lane&15)
  const int col = n0 + (lane & 15);
  const int rb = m0 + ((lane >> 4) << 3);
  const float bv = bias[col];
#pragma unroll
  for (int r = 0; r < 8; ++r) {
    float y = acc[r] + bv;
    if (ACT == 1) y = fmaxf(y, 0.f);
    if (ACT == 2) y = 1.f / (1.f + __expf(-y));
    const size_t o = (size_t)(rb + r) * N + col;
    if (Yh) Yh[o] = (_Float16)y;
    if (Yf) Yf[o] = y;
  }
}

// ---------------- conv1d k=11 SAME, tap-major WMMA (main path) -------------
// Y[b,o,w] = relu(bias[o] + sum_t sum_c In(b,c,w-5+t) * WgtT[t][o][c])
// K dimension = channels (C mult of 32); 11 shifted channel-GEMMs share one
// accumulator.  B-fragment elements sit at compile-time offsets pat(e)*80
// from a single per-lane LDS base -> immediate-offset ds_load_u16 gather.
// MODE 0: In = in0[b][CR][W].
// MODE 1: virtual concat: c<CH0 -> in0[b][CH0][W/8] nearest-upsampled x8,
//         else           -> in1[b][CR-CH0][W]  (decoder skip connection).
// Block: stage [C][74] halo tile in LDS (rows >= CR zeroed), 8 waves =
// 32(o) x 64(w) WMMA tiles.
template <int C, int CR, int MODE, int CH0>
__global__ __launch_bounds__(256) void conv11_tap_wmma(
    const _Float16* __restrict__ in0, const _Float16* __restrict__ in1,
    const _Float16* __restrict__ wgtT /*[11][O][C]*/,
    const float* __restrict__ bias, _Float16* __restrict__ out /*[B][O][W]*/,
    int O, int W) {
  __shared__ _Float16 smem[C * 80];
  const int tid = threadIdx.x;
  const int b = blockIdx.z;
  const int w0blk = blockIdx.x * 64;

  for (int idx = tid; idx < C * 74; idx += 256) {
    const int c = idx / 74;
    const int i = idx - c * 74;
    const int gw = w0blk - 5 + i;
    _Float16 v = (_Float16)0.f;
    if (c < CR && gw >= 0 && gw < W) {
      if (MODE == 0) {
        v = in0[((size_t)b * CR + c) * W + gw];
      } else {
        if (c < CH0) v = in0[((size_t)b * CH0 + c) * (W >> 3) + (gw >> 3)];
        else v = in1[((size_t)b * (CR - CH0) + (c - CH0)) * W + gw];
      }
    }
    smem[c * 80 + i] = v;
  }
  __syncthreads();

  const int lane = tid & 31;
  const int wave = tid >> 5;
  const int o0 = blockIdx.y * 32 + (wave & 1) * 16;
  const int n = (wave >> 1) * 16 + (lane & 15);  // local output position
  const int koff = (lane >> 4) << 3;
  const _Float16* bbase0 = smem + koff * 80 + n;

  v8f acc = {};
  for (int t = 0; t < 11; ++t) {
    const _Float16* Wr = wgtT + ((size_t)t * O + o0) * C;
    const _Float16* bb = bbase0 + t;
#pragma unroll 1
    for (int cb = 0; cb < C; cb += 32) {
      __builtin_prefetch(Wr + cb + 128, 0, 1);
      v16h a = frag_ld(Wr + cb, (size_t)C, lane);
      v16h bf;
#pragma unroll
      for (int e = 0; e < 16; ++e) {
        const int pe = ((e >> 3) << 4) + (e & 7);  // compile-time constant
        bf[e] = bb[pe * 80];                       // immediate DS offset
      }
      acc = wmma32f16(a, bf, acc);
      bb += 32 * 80;
    }
  }

  const int w = w0blk + n;
  if (w < W) {
    const int ob = o0 + ((lane >> 4) << 3);
#pragma unroll
    for (int r = 0; r < 8; ++r) {
      const int o = ob + r;
      const float y = fmaxf(acc[r] + bias[o], 0.f);
      out[((size_t)b * O + o) * W + w] = (_Float16)y;
    }
  }
}

// ------------- conv1d k=11 SAME, channel-tap K (conv1 only) ----------------
// Used only where C is tiny (conv1: C=2, K=22 padded to one 32-wide WMMA).
// The K loop runs exactly once, so the im2col cursors are computed once.
template <int C, int CP, int KPAD>
__global__ __launch_bounds__(256) void conv11_kk_wmma(
    const _Float16* __restrict__ in0,
    const _Float16* __restrict__ wgt /*[O][KPAD]*/,
    const float* __restrict__ bias, _Float16* __restrict__ out /*[B][O][W]*/,
    int O, int W) {
  __shared__ _Float16 smem[CP * 80];
  const int tid = threadIdx.x;
  const int b = blockIdx.z;
  const int w0blk = blockIdx.x * 64;

  for (int idx = tid; idx < CP * 74; idx += 256) {
    const int c = idx / 74;
    const int i = idx - c * 74;
    const int gw = w0blk - 5 + i;
    _Float16 v = (_Float16)0.f;
    if (c < C && gw >= 0 && gw < W) v = in0[((size_t)b * C + c) * W + gw];
    smem[c * 80 + i] = v;
  }
  __syncthreads();

  const int lane = tid & 31;
  const int wave = tid >> 5;
  const int o0 = blockIdx.y * 32 + (wave & 1) * 16;
  const int n = (wave >> 1) * 16 + (lane & 15);
  const int koff = (lane >> 4) << 3;
  const _Float16* Wr = wgt + (size_t)o0 * KPAD;

  v8f acc = {};
#pragma unroll
  for (int k0 = 0; k0 < KPAD; k0 += 32) {
    v16h a = frag_ld(Wr + k0, (size_t)KPAD, lane);
    v16h bf;
#pragma unroll
    for (int e = 0; e < 16; ++e) {
      const int kk = k0 + ((e >> 3) << 4) + koff + (e & 7);
      const int c = kk / 11;
      const int t = kk - c * 11;
      bf[e] = smem[c * 80 + n + t];
    }
    acc = wmma32f16(a, bf, acc);
  }

  const int w = w0blk + n;
  if (w < W) {
    const int ob = o0 + ((lane >> 4) << 3);
#pragma unroll
    for (int r = 0; r < 8; ++r) {
      const int o = ob + r;
      const float y = fmaxf(acc[r] + bias[o], 0.f);
      out[((size_t)b * O + o) * W + w] = (_Float16)y;
    }
  }
}

// ----------------------- small element-wise kernels ------------------------
__global__ void pool8_f16(const _Float16* __restrict__ in,
                          _Float16* __restrict__ out, int n) {
  const int i = blockIdx.x * blockDim.x + threadIdx.x;
  if (i >= n) return;
  const _Float16* p = in + (size_t)i * 8;
  float m = (float)p[0];
#pragma unroll
  for (int k = 1; k < 8; ++k) m = fmaxf(m, (float)p[k]);
  out[i] = (_Float16)m;
}

__global__ void cvt_f32_to_f16(const float* __restrict__ src,
                               _Float16* __restrict__ dst, int n) {
  const int i = blockIdx.x * blockDim.x + threadIdx.x;
  if (i < n) dst[i] = (_Float16)src[i];
}

// Convert fp32 weights [O][KREAL] -> f16 [O][KPAD] (zero pad K to mult of 32)
__global__ void cvt_pad_w(const float* __restrict__ src,
                          _Float16* __restrict__ dst, int O, int KREAL,
                          int KPAD) {
  const int i = blockIdx.x * blockDim.x + threadIdx.x;
  if (i >= O * KPAD) return;
  const int o = i / KPAD, k = i - o * KPAD;
  dst[i] = (k < KREAL) ? (_Float16)src[(size_t)o * KREAL + k] : (_Float16)0.f;
}

// Repack conv weights [O][CR][11] fp32 -> tap-major f16 [11][O][CP] (c pad 0)
__global__ void repack_w_tap(const float* __restrict__ src,
                             _Float16* __restrict__ dst, int O, int CR,
                             int CP) {
  const int i = blockIdx.x * blockDim.x + threadIdx.x;
  if (i >= 11 * O * CP) return;
  const int t = i / (O * CP);
  const int rem = i - t * O * CP;
  const int o = rem / CP;
  const int c = rem - o * CP;
  dst[i] = (c < CR) ? (_Float16)src[((size_t)o * CR + c) * 11 + t]
                    : (_Float16)0.f;
}

// ------------------ d_c1: 128ch -> 2ch conv, direct VALU -------------------
// Only 2 output channels (12% WMMA utilization) and memory-light: VALU path.
// Input = concat( repeat8(u1[b][64][160]), back1_other[b][64][1280] ).
__global__ __launch_bounds__(256) void dc1_conv(
    const _Float16* __restrict__ u1, const _Float16* __restrict__ back1,
    const float* __restrict__ wgt /*[2][128][11]*/,
    const float* __restrict__ bias, float* __restrict__ out /*[B][2][1280]*/) {
  const int i = blockIdx.x * 256 + threadIdx.x;  // exactly B*2*1280 threads
  const int w = i % 1280;
  const int o = (i / 1280) & 1;
  const int b = i / (2 * 1280);
  float acc = bias[o];
  const float* wr = wgt + (size_t)o * 128 * 11;
  const _Float16* ub = u1 + (size_t)b * 64 * 160;
  const _Float16* sb = back1 + (size_t)b * 64 * 1280;
  for (int c = 0; c < 64; ++c) {
    const float* wc0 = wr + c * 11;
    const float* wc1 = wr + (64 + c) * 11;
    const _Float16* uc = ub + (size_t)c * 160;
    const _Float16* sc = sb + (size_t)c * 1280;
#pragma unroll
    for (int t = 0; t < 11; ++t) {
      const int p = w - 5 + t;
      if (p >= 0 && p < 1280) {
        acc += (float)uc[p >> 3] * wc0[t];
        acc += (float)sc[p] * wc1[t];
      }
    }
  }
  out[i] = fmaxf(acc, 0.f);
}

// ------------------------------ pairwise KL --------------------------------
// kl = ((L-1)*sum_l A[l] - (sum C - tr C)) / (L*L/2) with A[l]==C[l,l]
//    = (L*tr(C) - sum(C)) / (L*L/2).   C[i,j] = mean_{m,d} logT[i]*T[j].
__global__ __launch_bounds__(256) void kl_pairC(
    const float* __restrict__ feas, const float* __restrict__ feat,
    const int* __restrict__ labs, const int* __restrict__ labt,
    float* __restrict__ Cmat /*[8][8]*/) {
  __shared__ int idxs[8][32];
  __shared__ int idxt[8][32];
  __shared__ float red[256];
  const int tid = threadIdx.x;
  if (tid == 0) {  // stable grouping (== boolean-mask gather order)
    int cs[8] = {0}, ct[8] = {0};
    for (int b = 0; b < 256; ++b) {
      const int ls = labs[b] & 7;
      idxs[ls][cs[ls]++ & 31] = b;
      const int lt = labt[b] & 7;
      idxt[lt][ct[lt]++ & 31] = b;
    }
  }
  __syncthreads();
  const int i = blockIdx.x >> 3;
  const int j = blockIdx.x & 7;
  float part = 0.f;
  for (int u = tid; u < 64 * 64; u += 256) {  // M=64 rows, D=64 features
    const int m = u >> 6;
    const int d = u & 63;
    const int bi = (m < 32) ? idxs[i][m] : idxt[i][m - 32];
    const int bj = (m < 32) ? idxs[j][m] : idxt[j][m - 32];
    const float ti = (m < 32) ? feas[bi * 64 + d] : feat[bi * 64 + d];
    const float tj = (m < 32) ? feas[bj * 64 + d] : feat[bj * 64 + d];
    part += __logf(ti) * tj;
  }
  red[tid] = part;
  __syncthreads();
  for (int s = 128; s > 0; s >>= 1) {
    if (tid < s) red[tid] += red[tid + s];
    __syncthreads();
  }
  if (tid == 0) Cmat[blockIdx.x] = red[0] / 4096.f;
}

__global__ void kl_final(const float* __restrict__ Cmat,
                         const int* __restrict__ plab,
                         float* __restrict__ outkl) {
  __shared__ float s_sum[64];
  __shared__ float s_tr[64];
  const int tid = threadIdx.x;
  const float c = Cmat[tid];
  s_sum[tid] = c;
  s_tr[tid] = ((tid >> 3) == (tid & 7)) ? c : 0.f;
  __syncthreads();
  for (int s = 32; s > 0; s >>= 1) {
    if (tid < s) {
      s_sum[tid] += s_sum[tid + s];
      s_tr[tid] += s_tr[tid + s];
    }
    __syncthreads();
  }
  if (tid == 0) {
    const float L = (float)(*plab);
    outkl[0] = (L * s_tr[0] - s_sum[0]) / (L * L * 0.5f);
  }
}

// ---------------------------------------------------------------------------
extern "C" void kernel_launch(void* const* d_in, const int* in_sizes, int n_in,
                              void* d_out, int out_size, void* d_ws,
                              size_t ws_size, hipStream_t stream) {
  (void)in_sizes; (void)n_in; (void)out_size; (void)ws_size;
  const float* x_s = (const float*)d_in[0];
  const float* x_t = (const float*)d_in[1];
  const int* lab_s = (const int*)d_in[2];
  const int* lab_t = (const int*)d_in[3];
  const int* lab_all = (const int*)d_in[4];
  const float* e_c1_w = (const float*)d_in[5];
  const float* e_c1_b = (const float*)d_in[6];
  const float* e_c2_w = (const float*)d_in[7];
  const float* e_c2_b = (const float*)d_in[8];
  const float* e_f1_w = (const float*)d_in[9];
  const float* e_f1_b = (const float*)d_in[10];
  const float* e_f2_w = (const float*)d_in[11];
  const float* e_f2_b = (const float*)d_in[12];
  const float* d_f1_w = (const float*)d_in[13];
  const float* d_f1_b = (const float*)d_in[14];
  const float* d_f2_w = (const float*)d_in[15];
  const float* d_f2_b = (const float*)d_in[16];
  const float* d_c2_w = (const float*)d_in[17];
  const float* d_c2_b = (const float*)d_in[18];
  const float* d_c1_w = (const float*)d_in[19];
  const float* d_c1_b = (const float*)d_in[20];
  float* out = (float*)d_out;

  const int B = 256;
  char* base = (char*)d_ws;
  size_t off = 0;
  auto alloc = [&](size_t bytes) -> void* {
    void* p = base + off;
    off = (off + bytes + 255) & ~(size_t)255;
    return p;
  };

  // f16 weights
  _Float16* w1p = (_Float16*)alloc((size_t)64 * 32 * 2);          // [O][KPAD]
  _Float16* w2t = (_Float16*)alloc((size_t)11 * 128 * 64 * 2);    // [11][O][C]
  _Float16* wf1 = (_Float16*)alloc((size_t)512 * 2560 * 2);
  _Float16* wf2 = (_Float16*)alloc((size_t)64 * 512 * 2);
  _Float16* wd1 = (_Float16*)alloc((size_t)256 * 64 * 2);
  _Float16* wd2 = (_Float16*)alloc((size_t)2560 * 256 * 2);
  _Float16* wc2t = (_Float16*)alloc((size_t)11 * 64 * 256 * 2);   // [11][O][C]
  // per-direction activations (0 = source, 1 = target)
  _Float16 *xh[2], *bk1[2], *h1[2], *bk2[2], *h2[2], *a1[2], *feah[2];
  float* fea32[2];
  for (int d = 0; d < 2; ++d) {
    xh[d] = (_Float16*)alloc((size_t)B * 2 * 1280 * 2);
    bk1[d] = (_Float16*)alloc((size_t)B * 64 * 1280 * 2);
    h1[d] = (_Float16*)alloc((size_t)B * 64 * 160 * 2);
    bk2[d] = (_Float16*)alloc((size_t)B * 128 * 160 * 2);
    h2[d] = (_Float16*)alloc((size_t)B * 128 * 20 * 2);
    a1[d] = (_Float16*)alloc((size_t)B * 512 * 2);
    feah[d] = (_Float16*)alloc((size_t)B * 64 * 2);
    fea32[d] = (float*)alloc((size_t)B * 64 * 4);
  }
  _Float16* g1 = (_Float16*)alloc((size_t)B * 256 * 2);
  _Float16* g2 = (_Float16*)alloc((size_t)B * 2560 * 2);
  _Float16* u1 = (_Float16*)alloc((size_t)B * 64 * 160 * 2);
  float* Cmat = (float*)alloc(64 * 4);

  auto g1d = [](int n) { return dim3((unsigned)((n + 255) / 256)); };

  // weight + input conversion
  cvt_pad_w<<<g1d(64 * 32), 256, 0, stream>>>(e_c1_w, w1p, 64, 22, 32);
  repack_w_tap<<<g1d(11 * 128 * 64), 256, 0, stream>>>(e_c2_w, w2t, 128, 64, 64);
  cvt_pad_w<<<g1d(512 * 2560), 256, 0, stream>>>(e_f1_w, wf1, 512, 2560, 2560);
  cvt_pad_w<<<g1d(64 * 512), 256, 0, stream>>>(e_f2_w, wf2, 64, 512, 512);
  cvt_pad_w<<<g1d(256 * 64), 256, 0, stream>>>(d_f1_w, wd1, 256, 64, 64);
  cvt_pad_w<<<g1d(2560 * 256), 256, 0, stream>>>(d_f2_w, wd2, 2560, 256, 256);
  repack_w_tap<<<g1d(11 * 64 * 256), 256, 0, stream>>>(d_c2_w, wc2t, 64, 256, 256);
  cvt_f32_to_f16<<<g1d(B * 2 * 1280), 256, 0, stream>>>(x_s, xh[0], B * 2 * 1280);
  cvt_f32_to_f16<<<g1d(B * 2 * 1280), 256, 0, stream>>>(x_t, xh[1], B * 2 * 1280);

  // encoders (source, target)
  for (int d = 0; d < 2; ++d) {
    conv11_kk_wmma<2, 3, 32><<<dim3(20, 2, B), 256, 0, stream>>>(
        xh[d], w1p, e_c1_b, bk1[d], 64, 1280);
    pool8_f16<<<g1d(B * 64 * 160), 256, 0, stream>>>(bk1[d], h1[d], B * 64 * 160);
    conv11_tap_wmma<64, 64, 0, 0><<<dim3(3, 4, B), 256, 0, stream>>>(
        h1[d], nullptr, w2t, e_c2_b, bk2[d], 128, 160);
    pool8_f16<<<g1d(B * 128 * 20), 256, 0, stream>>>(bk2[d], h2[d], B * 128 * 20);
    fc_wmma<1><<<dim3(512 / 64, B / 32), 256, 0, stream>>>(
        h2[d], wf1, e_f1_b, a1[d], nullptr, B, 512, 2560);
    fc_wmma<2><<<dim3(1, B / 32), 256, 0, stream>>>(
        a1[d], wf2, e_f2_b, feah[d], fea32[d], B, 64, 512);
  }

  // decoders with crossed skip connections: out_s = dec(fea_s, back_t)
  for (int o = 0; o < 2; ++o) {
    const int f = o, r = 1 - o;
    fc_wmma<1><<<dim3(256 / 64, B / 32), 256, 0, stream>>>(
        feah[f], wd1, d_f1_b, g1, nullptr, B, 256, 64);
    fc_wmma<1><<<dim3(2560 / 64, B / 32), 256, 0, stream>>>(
        g1, wd2, d_f2_b, g2, nullptr, B, 2560, 256);
    conv11_tap_wmma<256, 256, 1, 128><<<dim3(3, 2, B), 256, 0, stream>>>(
        g2, bk2[r], wc2t, d_c2_b, u1, 64, 160);
    dc1_conv<<<dim3((B * 2 * 1280) / 256), 256, 0, stream>>>(
        u1, bk1[r], d_c1_w, d_c1_b, out + (size_t)o * B * 2 * 1280);
  }

  // pairwise KL -> d_out tail
  kl_pairC<<<64, 256, 0, stream>>>(fea32[0], fea32[1], lab_s, lab_t, Cmat);
  kl_final<<<1, 64, 0, stream>>>(Cmat, lab_all, out + (size_t)2 * B * 2 * 1280);
}
